// MPMesh_30253749633773
// MI455X (gfx1250) — matrix operand order repaired and checked by reading, not compiled, not run on hardware
//
#include <hip/hip_runtime.h>
#include <math.h>

typedef __attribute__((ext_vector_type(16))) _Float16 v16h;
typedef __attribute__((ext_vector_type(8)))  _Float16 v8h;
typedef __attribute__((ext_vector_type(8)))  float    v8f;

#define IMG_H 240
#define IMG_W 320
#define MPI_HH 300
#define MPI_WW 400
#define MPI_DD 24
#define G_W 6
#define AT_H 1200
#define AT_W 2400
#define WPB 4          // waves per block (128 threads)

__device__ __forceinline__ float sigmoid_f(float x) {
    return 1.0f / (1.0f + __expf(-x));
}

__global__ __launch_bounds__(WPB * 32) void mpi_render_wmma(
    const float* __restrict__ extrin,   // (4,4)
    const float* __restrict__ intrin,   // (3,3)
    const float* __restrict__ atlas,    // (16,1200,2400)
    const float* __restrict__ w1,       // (43,48)
    const float* __restrict__ b1,       // (48)
    const float* __restrict__ w2,       // (48,4)
    const float* __restrict__ b2,       // (4)
    float* __restrict__ out)            // (240,320,3)
{
    __shared__ __align__(16) _Float16 feat_lds[WPB][16][64]; // x tile (16 samples x 64, K-pad)
    __shared__ __align__(16) _Float16 h_lds[WPB][16][64];    // hidden tile (16 x 64, K-pad)
    __shared__ __align__(16) float    rgba_lds[WPB][16][4];

    const int lane = threadIdx.x & 31;
    const int wv   = threadIdx.x >> 5;
    const int hf   = lane >> 4;    // wave half
    const int lm   = lane & 15;

    _Float16 (*feat)[64] = feat_lds[wv];
    _Float16 (*hbuf)[64] = h_lds[wv];
    float    (*rgba)[4]  = rgba_lds[wv];

    // ---- camera setup (uniform across lanes) ----
    float R[3][3], T[3];
    #pragma unroll
    for (int r = 0; r < 3; ++r) {
        #pragma unroll
        for (int c = 0; c < 3; ++c) R[r][c] = extrin[r * 4 + c];
        T[r] = extrin[r * 4 + 3];
    }
    const float fx = intrin[0], cxk = intrin[2];
    const float fy = intrin[4], cyk = intrin[5];
    float cam_o[3];
    #pragma unroll
    for (int i = 0; i < 3; ++i)
        cam_o[i] = -(R[0][i] * T[0] + R[1][i] * T[1] + R[2][i] * T[2]);

    // ---- stage MLP weights as WMMA B operands (persistent in VGPRs) ----
    // Biases are FOLDED into the GEMMs via constant-1 feature columns:
    //   input col 43 == 1, B1 row K=43 holds b1[n]
    //   hidden col 48 == 1, B2 row K=48 holds b2[n]
    // B staged in contiguous K order (32x16): lane holds column n = lm; element e -> K = hf*16 + e
    v16h B1[6];   // [kt*3 + nt], kt in {0,1} (K 0-31 / 32-63), nt in {0,1,2} (N 0-47)
    #pragma unroll
    for (int kt = 0; kt < 2; ++kt) {
        #pragma unroll
        for (int nt = 0; nt < 3; ++nt) {
            v16h b;
            const int n = nt * 16 + lm;
            #pragma unroll
            for (int e = 0; e < 16; ++e) {
                const int K = kt * 32 + hf * 16 + e;
                float v;
                if (K < 43)       v = w1[K * 48 + n];
                else if (K == 43) v = b1[n];           // bias row
                else              v = 0.0f;
                b[e] = (_Float16)v;
            }
            B1[kt * 3 + nt] = b;
        }
    }
    v16h B2[2];
    #pragma unroll
    for (int kt = 0; kt < 2; ++kt) {
        v16h b;
        const int n = lm;
        #pragma unroll
        for (int e = 0; e < 16; ++e) {
            const int K = kt * 32 + hf * 16 + e;
            float v = 0.0f;
            if (n < 4) {
                if (K < 48)       v = w2[K * 4 + n];
                else if (K == 48) v = b2[n];           // bias row
            }
            b[e] = (_Float16)v;
        }
        B2[kt] = b;
    }

    // ---- per-pixel geometry (both wave halves compute pixel lm) ----
    const int pbase = (blockIdx.x * WPB + wv) * 16;
    const int p     = pbase + lm;
    const int piy   = p / IMG_W;
    const int pix   = p - piy * IMG_W;

    const float dc0 = ((float)pix - cxk) / fx;
    const float dc1 = ((float)piy - cyk) / fy;
    float dw[3];
    #pragma unroll
    for (int i = 0; i < 3; ++i)
        dw[i] = R[0][i] * dc0 + R[1][i] * dc1 + R[2][i] * 1.0f;
    const float inv_n = rsqrtf(dw[0] * dw[0] + dw[1] * dw[1] + dw[2] * dw[2]);

    // view embedding (d-independent): 27 values
    float ve[27];
    #pragma unroll
    for (int i = 0; i < 3; ++i) ve[i] = dw[i] * inv_n;
    #pragma unroll
    for (int k = 0; k < 4; ++k) {
        const float s = (float)(1 << k);
        #pragma unroll
        for (int i = 0; i < 3; ++i) {
            ve[3 + 6 * k + i] = sinf(ve[i] * s);
            ve[6 + 6 * k + i] = cosf(ve[i] * s);
        }
    }

    // ---- one-time LDS init: bias-1 columns, K-pad zeros, view embedding ----
    if (lane < 16) {
        feat[lm][43] = (_Float16)1.0f;                         // bias feature
        for (int k = 44; k < 64; ++k) feat[lm][k] = (_Float16)0.0f;
        hbuf[lm][48] = (_Float16)1.0f;                         // bias feature
        for (int k = 49; k < 64; ++k) hbuf[lm][k] = (_Float16)0.0f;
    }
    #pragma unroll
    for (int i = 0; i < 14; ++i) {
        const int idx = hf * 13 + i;        // hf0: 0..13, hf1: 13..26
        if (idx < 27) feat[lm][16 + idx] = (_Float16)ve[idx];
    }
    __builtin_amdgcn_wave_barrier();

    // ---- plane loop: near (d=23) to far (d=0), front-to-back compositing ----
    float outr = 0.0f, outg = 0.0f, outb = 0.0f, Tacc = 1.0f;
    const v8f vzero = {0.f, 0.f, 0.f, 0.f, 0.f, 0.f, 0.f, 0.f};

    for (int d = MPI_DD - 1; d >= 0; --d) {
        // depths = (1/linspace(1/NEAR,1/FAR,24))[::-1]
        const float invd  = 0.5f + (0.01f - 0.5f) * (float)(23 - d) * (1.0f / 23.0f);
        const float depth = 1.0f / invd;
        const float t     = (depth - cam_o[2]) / dw[2];
        const float ptx   = cam_o[0] + t * dw[0];
        const float pty   = cam_o[1] + t * dw[1];
        const float ptz   = depth;                       // cam_o.z + t*dw.z exactly
        const float pxm   = ptx / ptz * 300.0f + 200.0f; // REF_FX, cx_mpi = 160+40
        const float pym   = pty / ptz * 300.0f + 150.0f; // REF_FY, cy_mpi = 120+30
        const bool valid  = (t > 0.0f) && (pxm >= 0.0f) && (pxm <= (float)(MPI_WW - 1)) &&
                            (pym >= 0.0f) && (pym <= (float)(MPI_HH - 1));
        const float pxc = fminf(fmaxf(pxm, 0.0f), (float)(MPI_WW - 1));
        const float pyc = fminf(fmaxf(pym, 0.0f), (float)(MPI_HH - 1));

        const float col = (float)(d % G_W);
        const float row = (float)(d / G_W);
        const float ax  = (col + pxc * (1.0f / 399.0f)) * ((float)(AT_W - 1) / 6.0f);
        const float ay  = (row + pyc * (1.0f / 299.0f)) * ((float)(AT_H - 1) / 4.0f);
        const int x0 = (int)floorf(ax);
        const int y0 = (int)floorf(ay);
        const int x1 = (x0 + 1 < AT_W) ? x0 + 1 : AT_W - 1;
        const int y1 = (y0 + 1 < AT_H) ? y0 + 1 : AT_H - 1;
        const float wx = ax - (float)x0;
        const float wy = ay - (float)y0;
        const float w00 = (1.0f - wx) * (1.0f - wy);
        const float w01 = wx * (1.0f - wy);
        const float w10 = (1.0f - wx) * wy;
        const float w11 = wx * wy;
        const size_t o00 = (size_t)y0 * AT_W + x0;
        const size_t o01 = (size_t)y0 * AT_W + x1;
        const size_t o10 = (size_t)y1 * AT_W + x0;
        const size_t o11 = (size_t)y1 * AT_W + x1;

        // each wave-half gathers 8 of the 16 atlas channels for its pixel
        #pragma unroll
        for (int c = 0; c < 8; ++c) {
            const int ch = hf * 8 + c;
            const float* chp = atlas + (size_t)ch * ((size_t)AT_H * AT_W);
            const float fv = chp[o00] * w00 + chp[o01] * w01 +
                             chp[o10] * w10 + chp[o11] * w11;
            feat[lm][hf * 8 + c] = (_Float16)fv;
        }
        __builtin_amdgcn_wave_barrier();

        // ---- layer 1: (16x64) x (64x48) via 2x3 WMMA f32_16x16x32_f16 ----
        v8f acc0 = vzero, acc1 = vzero, acc2 = vzero;
        #pragma unroll
        for (int kb = 0; kb < 2; ++kb) {
            // A staged in contiguous K order (16x32): row = lm; elems 0-15 = K off..off+15,
            // off = kb*32 + hf*16
            v16h a;
            const _Float16* src = &feat[lm][kb * 32 + hf * 16];
            ((v8h*)&a)[0] = *(const v8h*)(src);
            ((v8h*)&a)[1] = *(const v8h*)(src + 8);
            acc0 = __builtin_amdgcn_wmma_f32_16x16x32_f16(false, a, false, B1[kb * 3 + 0],
                                                          (short)0, acc0, false, false);
            acc1 = __builtin_amdgcn_wmma_f32_16x16x32_f16(false, a, false, B1[kb * 3 + 1],
                                                          (short)0, acc1, false, false);
            acc2 = __builtin_amdgcn_wmma_f32_16x16x32_f16(false, a, false, B1[kb * 3 + 2],
                                                          (short)0, acc2, false, false);
        }

        // ReLU only (bias already folded), transpose through LDS for layer 2.
        // C layout: lane holds column n = nt*16+lm; VGPR r -> sample m = r + hf*8
        #pragma unroll
        for (int nt = 0; nt < 3; ++nt) {
            const v8f acc = (nt == 0) ? acc0 : ((nt == 1) ? acc1 : acc2);
            const int n = nt * 16 + lm;
            #pragma unroll
            for (int r = 0; r < 8; ++r) {
                hbuf[r + hf * 8][n] = (_Float16)fmaxf(acc[r], 0.0f);
            }
        }
        __builtin_amdgcn_wave_barrier();

        // ---- layer 2: (16x64) x (64x16) via 2 WMMA ----
        v8f accO = vzero;
        #pragma unroll
        for (int kb = 0; kb < 2; ++kb) {
            v16h a;
            const _Float16* src = &hbuf[lm][kb * 32 + hf * 16];
            ((v8h*)&a)[0] = *(const v8h*)(src);
            ((v8h*)&a)[1] = *(const v8h*)(src + 8);
            accO = __builtin_amdgcn_wmma_f32_16x16x32_f16(false, a, false, B2[kb],
                                                          (short)0, accO, false, false);
        }

        // sigmoid only (bias already folded) + scatter rgba (channels 0..3)
        if (lm < 4) {
            #pragma unroll
            for (int r = 0; r < 8; ++r) {
                rgba[r + hf * 8][lm] = sigmoid_f(accO[r]);
            }
        }
        __builtin_amdgcn_wave_barrier();

        // front-to-back composite (lanes 0..15, one pixel each)
        if (lane < 16) {
            const float a = valid ? rgba[lm][3] : 0.0f;
            const float w = a * Tacc;
            outr += rgba[lm][0] * w;
            outg += rgba[lm][1] * w;
            outb += rgba[lm][2] * w;
            Tacc *= (1.0f - a);
        }
        __builtin_amdgcn_wave_barrier();
    }

    if (lane < 16) {
        out[(size_t)p * 3 + 0] = outr;
        out[(size_t)p * 3 + 1] = outg;
        out[(size_t)p * 3 + 2] = outb;
    }
}

extern "C" void kernel_launch(void* const* d_in, const int* in_sizes, int n_in,
                              void* d_out, int out_size, void* d_ws, size_t ws_size,
                              hipStream_t stream) {
    (void)in_sizes; (void)n_in; (void)out_size; (void)d_ws; (void)ws_size;
    // setup_inputs order: H, W, extrin, intrin, atlas, w1, b1, w2, b2
    const float* extrin = (const float*)d_in[2];
    const float* intrin = (const float*)d_in[3];
    const float* atlas  = (const float*)d_in[4];
    const float* w1     = (const float*)d_in[5];
    const float* b1     = (const float*)d_in[6];
    const float* w2     = (const float*)d_in[7];
    const float* b2     = (const float*)d_in[8];
    float* out = (float*)d_out;

    const int total_waves = (IMG_H * IMG_W) / 16;   // 4800 (exact)
    const int blocks      = total_waves / WPB;      // 1200
    mpi_render_wmma<<<blocks, WPB * 32, 0, stream>>>(extrin, intrin, atlas,
                                                     w1, b1, w2, b2, out);
}
